// Model_24292335026623
// MI455X (gfx1250) — compile-verified, matrix operand order checked by reading, not run on hardware
//
#include <hip/hip_runtime.h>
#include <hip/hip_bf16.h>

// ---------------------------------------------------------------------------
// S4D-DEQ siamese encoder for MI455X (gfx1250, wave32, WMMA + TDM).
// Layouts are channel-major [b][h][l] throughout so the SSM scan, LN, GEMM
// B-tiles and pooling are all contiguous in l. GEMMs use
// v_wmma_f32_16x16x32_bf16 (bf16 in, f32 accumulate); the weight tile is
// DMA'd into LDS by the Tensor Data Mover (tensor_load_to_lds + TENSORcnt).
// ---------------------------------------------------------------------------

typedef __bf16 bf16x8  __attribute__((ext_vector_type(8)));
typedef __bf16 bf16x16 __attribute__((ext_vector_type(16)));
typedef float  v8f     __attribute__((ext_vector_type(8)));
typedef unsigned int v4u __attribute__((ext_vector_type(4)));
typedef int          v8i __attribute__((ext_vector_type(8)));
typedef int          v4i __attribute__((ext_vector_type(4)));

#define H_MODEL 256
#define N_STATE 64
#define SEQ     2048
#define BB      32                       // 2 encodes x 16 batch fused
#define NELEM   ((size_t)BB * H_MODEL * SEQ)   // 16,777,216 floats per tensor

__device__ __forceinline__ unsigned short f2bf(float f) {
    unsigned int u = __float_as_uint(f);
    u += 0x7fffu + ((u >> 16) & 1u);     // round-to-nearest-even
    return (unsigned short)(u >> 16);
}
__device__ __forceinline__ float bf2f(unsigned int u) {
    return __uint_as_float(u << 16);
}

// --------------------------- small prep kernels ----------------------------

__global__ void kInit(float* acc) {
    if (threadIdx.x < 4) acc[threadIdx.x] = 0.f;
}

// lambda = exp(dt*A), Ceff = C * (exp(dt*A)-1)/A   (per h,n)  -> float4
__global__ void kPrepParams(const float* __restrict__ logdt,
                            const float* __restrict__ are, const float* __restrict__ aim,
                            const float* __restrict__ cre, const float* __restrict__ cim,
                            float4* __restrict__ dp) {
    int g = blockIdx.x * 256 + threadIdx.x;         // 16384 = 256*64
    int h = g >> 6, n = g & 63;
    float dt = expf(logdt[h]);
    float ar = are[h * 64 + n], ai = aim[h * 64 + n];
    float cr = cre[h * 64 + n], ci = cim[h * 64 + n];
    float er = expf(ar * dt);
    float lr = er * cosf(ai * dt);
    float li = er * sinf(ai * dt);
    float e1r = lr - 1.f, e1i = li;
    float den = 1.f / (ar * ar + ai * ai);
    float qr = (e1r * ar + e1i * ai) * den;
    float qi = (e1i * ar - e1r * ai) * den;
    float4 o;
    o.x = lr; o.y = li;
    o.z = cr * qr - ci * qi;
    o.w = cr * qi + ci * qr;
    dp[g] = o;
}

// WoT[i][j] = bf16(Wo[j][i])
__global__ void kPrepW(const float* __restrict__ Wo, unsigned short* __restrict__ WT) {
    int j = blockIdx.x, i = threadIdx.x;
    WT[i * 256 + j] = f2bf(Wo[j * 256 + i]);
}

// ----------------------- embedding + LN (transposed out) -------------------

__global__ void kEmbedLN(const int* __restrict__ x1, const int* __restrict__ x2,
                         const float* __restrict__ emb,
                         const float* __restrict__ lnw, const float* __restrict__ lnb,
                         float* __restrict__ Hb) {
    __shared__ float tile[32][257];
    int b = blockIdx.y, l0 = blockIdx.x * 32, tid = threadIdx.x;
    const int* T = (b >= 16) ? x2 : x1;
    int bb = b & 15;
    for (int r = 0; r < 32; r++) {
        int tok = T[bb * SEQ + l0 + r];
        tile[r][tid] = emb[tok * H_MODEL + tid];
    }
    __syncthreads();
    int w = tid >> 5, lane = tid & 31;
    for (int rr = 0; rr < 4; rr++) {
        int r = w * 4 + rr;
        float sx = 0.f, sxx = 0.f;
        for (int k = 0; k < 8; k++) { float x = tile[r][lane + 32 * k]; sx += x; sxx += x * x; }
        for (int off = 16; off; off >>= 1) { sx += __shfl_xor(sx, off); sxx += __shfl_xor(sxx, off); }
        float m = sx * (1.f / 256.f);
        float v = sxx * (1.f / 256.f) - m * m;
        float rs = rsqrtf(v + 1e-5f);
        for (int k = 0; k < 8; k++) {
            int c = lane + 32 * k;
            tile[r][c] = (tile[r][c] - m) * rs * lnw[c] + lnb[c];
        }
    }
    __syncthreads();
    float* dst = Hb + ((size_t)b * H_MODEL + tid) * SEQ + l0;
    for (int rq = 0; rq < 8; rq++) {
        float4 o;
        o.x = tile[rq * 4 + 0][tid]; o.y = tile[rq * 4 + 1][tid];
        o.z = tile[rq * 4 + 2][tid]; o.w = tile[rq * 4 + 3][tid];
        *(float4*)(dst + rq * 4) = o;
    }
}

// ------------------ pre-block LayerNorm over h (channel-major) -------------

__global__ void kLN(const float* __restrict__ Hb,
                    const float* __restrict__ lnw, const float* __restrict__ lnb,
                    float* __restrict__ Z) {
    int b = blockIdx.y;
    int l = blockIdx.x * 256 + threadIdx.x;
    const float* src = Hb + (size_t)b * H_MODEL * SEQ + l;
    float sx = 0.f, sxx = 0.f;
    for (int h = 0; h < H_MODEL; h++) { float x = src[(size_t)h * SEQ]; sx += x; sxx += x * x; }
    float m = sx * (1.f / 256.f);
    float v = sxx * (1.f / 256.f) - m * m;
    float rs = rsqrtf(v + 1e-5f);
    float* dst = Z + (size_t)b * H_MODEL * SEQ + l;
    for (int h = 0; h < H_MODEL; h++) {
        float x = src[(size_t)h * SEQ];
        dst[(size_t)h * SEQ] = (x - m) * rs * lnw[h] + lnb[h];
    }
}

// ----------------- diagonal SSM scan: 4 lanes x 16 states / channel --------

__global__ void kScan(const float4* __restrict__ dp, const float* __restrict__ Z,
                      unsigned short* __restrict__ Y, const float* __restrict__ Dv) {
    int tid = threadIdx.x;
    int g = tid >> 2, j = tid & 3;                  // 4-lane group per (b,h)
    int c = blockIdx.x * 64 + g;                    // 8192 channels total
    int b = c >> 8, h = c & 255;

    const float4* dpp = dp + h * N_STATE + j * 16;
    float4 lc[16];
#pragma unroll
    for (int q = 0; q < 16; q++) lc[q] = dpp[q];
    float sRe[16] = {0.f}, sIm[16] = {0.f};
    float Dh = Dv[h];

    const float4* zp = (const float4*)(Z + ((size_t)b * H_MODEL + h) * SEQ);
    unsigned short* yp = Y + ((size_t)b * H_MODEL + h) * SEQ;

    float keep = 0.f;
    for (int l4 = 0; l4 < SEQ / 4; l4++) {
        float4 u4 = zp[l4];
        float uu[4] = {u4.x, u4.y, u4.z, u4.w};
#pragma unroll
        for (int t = 0; t < 4; t++) {
            float u = uu[t];
            float ys = 0.f;
#pragma unroll
            for (int q = 0; q < 16; q++) {
                float oR = sRe[q], oI = sIm[q];
                float nr = fmaf(lc[q].x, oR, fmaf(-lc[q].y, oI, u));
                float ni = fmaf(lc[q].x, oI, lc[q].y * oR);
                sRe[q] = nr; sIm[q] = ni;
                ys = fmaf(lc[q].z, nr, ys);
                ys = fmaf(-lc[q].w, ni, ys);
            }
            ys += __shfl_xor(ys, 1);                // reduce across the 4-lane group
            ys += __shfl_xor(ys, 2);
            float out = tanhf(2.f * ys + Dh * u);
            if (j == t) keep = out;
        }
        yp[l4 * 4 + j] = f2bf(keep);
    }
}

// -------------------- H += Y @ Wo + bo   (bf16 WMMA GEMM) ------------------
// A = WoT [i][k] bf16 staged by the Tensor Data Mover (TDM) into LDS with
// row padding (64B data + 16B pad = 80B pitch). B = Y [b][k][l] bf16 staged
// synchronously with transpose. C/D f32. WG tile 128x128, 8 waves of
// 32(M)x64(N) -> 2x4 WMMA fragments each.

__global__ __launch_bounds__(256) void kGemm(const unsigned short* __restrict__ WT,
                                             const unsigned short* __restrict__ Y,
                                             float* __restrict__ Hb,
                                             const float* __restrict__ bo) {
    __shared__ __align__(16) unsigned short sA[128][40];   // [i][k] via TDM, 80B pitch
    __shared__ __align__(16) unsigned short sB[128][40];   // [n][k]

    const int t  = threadIdx.x;
    const int b  = blockIdx.z;
    const int i0 = blockIdx.y * 128;
    const int l0 = blockIdx.x * 128;
    const int w = t >> 5, lane = t & 31;
    const int wm = w >> 1, wn = w & 1;
    const int half = lane >> 4, lm = lane & 15, kk = half * 8;

    v8f acc[2][4];
    const v8f z8 = {0.f, 0.f, 0.f, 0.f, 0.f, 0.f, 0.f, 0.f};
#pragma unroll
    for (int mf = 0; mf < 2; mf++)
#pragma unroll
        for (int nf = 0; nf < 4; nf++) acc[mf][nf] = z8;

    const int kb = t >> 3, nc = t & 7;              // B staging: 1 row x 8n per thread
    const unsigned short* Brow = Y + ((size_t)b * 256 + kb) * 2048 + l0 + nc * 8;

    for (int ks = 0; ks < 8; ks++) {
        const int k0 = ks * 32;
        // ---- A tile via Tensor Data Mover: 128 rows x 32 bf16, stride 256,
        //      LDS gets 16B pad after each 64B row -> pitch 80B (sA[.][40]).
        if (t < 32) {                                // wave 0 issues the DMA
            unsigned long long ga =
                (unsigned long long)(uintptr_t)(WT + (size_t)(i0)*256 + k0);
            unsigned ldsa = (unsigned)(uintptr_t)&sA[0][0];
            v4u g0;
            g0.x = 1u;                                            // count=1 (valid)
            g0.y = ldsa;                                          // lds_addr
            g0.z = (unsigned)(ga & 0xffffffffu);                  // global_addr lo
            g0.w = (unsigned)((ga >> 32) & 0x01ffffffu) | (2u << 30);  // hi | type=2
            v8i g1;
            g1[0] = (1 << 16)        // data_size = 2 bytes
                  | (1 << 20)        // pad_enable
                  | (3 << 22)        // pad_interval: 16 dwords (64B row)
                  | (3 << 25);       // pad_amount: 4 dwords (16B)
            g1[1] = (256 & 0xffff) << 16;   // tensor_dim0 = 256 (bits 63:48)
            g1[2] = (256 & 0xffff) << 16;   // dim0 hi=0 | tensor_dim1 = 256
            g1[3] = (32  & 0xffff) << 16;   // dim1 hi=0 | tile_dim0 = 32
            g1[4] = 128;                    // tile_dim1 = 128, tile_dim2 = 0
            g1[5] = 256;                    // tensor_dim0_stride lo = 256
            g1[6] = 0;
            g1[7] = 0;
            v4i g2 = {0, 0, 0, 0};          // 2-D tile: groups 2/3 unused
            v4i g3 = {0, 0, 0, 0};
            v8i g4 = {0, 0, 0, 0, 0, 0, 0, 0};   // extra group (6-arg toolchain form)
            __builtin_amdgcn_tensor_load_to_lds(g0, g1, g2, g3, g4, 0);
            __builtin_amdgcn_s_wait_tensorcnt(0);
        }
        // ---- B tile staging (transpose to [n][k])
        uint4 q = *(const uint4*)(Brow + (size_t)k0 * 2048);
        sB[nc * 8 + 0][kb] = (unsigned short)(q.x & 0xffffu);
        sB[nc * 8 + 1][kb] = (unsigned short)(q.x >> 16);
        sB[nc * 8 + 2][kb] = (unsigned short)(q.y & 0xffffu);
        sB[nc * 8 + 3][kb] = (unsigned short)(q.y >> 16);
        sB[nc * 8 + 4][kb] = (unsigned short)(q.z & 0xffffu);
        sB[nc * 8 + 5][kb] = (unsigned short)(q.z >> 16);
        sB[nc * 8 + 6][kb] = (unsigned short)(q.w & 0xffffu);
        sB[nc * 8 + 7][kb] = (unsigned short)(q.w >> 16);
        if (ks < 7)
            __builtin_prefetch(Brow + (size_t)(k0 + 32) * 2048, 0, 1);  // global_prefetch_b8
        __syncthreads();

        bf16x16 af[2], bfr[4];
#pragma unroll
        for (int mf = 0; mf < 2; mf++) {
            int i = wm * 32 + mf * 16 + lm;
            bf16x8 lo = *(const bf16x8*)&sA[i][kk];
            bf16x8 hi = *(const bf16x8*)&sA[i][kk + 16];
            af[mf] = __builtin_shufflevector(lo, hi, 0, 1, 2, 3, 4, 5, 6, 7,
                                             8, 9, 10, 11, 12, 13, 14, 15);
        }
#pragma unroll
        for (int nf = 0; nf < 4; nf++) {
            int n = wn * 64 + nf * 16 + lm;
            bf16x8 lo = *(const bf16x8*)&sB[n][kk];
            bf16x8 hi = *(const bf16x8*)&sB[n][kk + 16];
            bfr[nf] = __builtin_shufflevector(lo, hi, 0, 1, 2, 3, 4, 5, 6, 7,
                                              8, 9, 10, 11, 12, 13, 14, 15);
        }
#pragma unroll
        for (int mf = 0; mf < 2; mf++)
#pragma unroll
            for (int nf = 0; nf < 4; nf++)
                acc[mf][nf] = __builtin_amdgcn_wmma_f32_16x16x32_bf16(
                    false, af[mf], false, bfr[nf], (short)0, acc[mf][nf], false, false);
        __syncthreads();
    }

#pragma unroll
    for (int mf = 0; mf < 2; mf++) {
#pragma unroll
        for (int r = 0; r < 8; r++) {
            int m = wm * 32 + mf * 16 + r + half * 8;
            float bias = bo[i0 + m];
#pragma unroll
            for (int nf = 0; nf < 4; nf++) {
                int n = wn * 64 + nf * 16 + lm;
                size_t idx = ((size_t)b * 256 + i0 + m) * 2048 + l0 + n;
                Hb[idx] += acc[mf][nf][r] + bias;
            }
        }
    }
}

// ---------------------------- DEQ elementwise ------------------------------

__global__ void kReluAdd(float* __restrict__ Hb, const float* __restrict__ XD,
                         unsigned short* __restrict__ ZP, int wz) {
    size_t e = (size_t)blockIdx.x * 256 + threadIdx.x;   // float4 index
    float4 h = ((const float4*)Hb)[e];
    float4 r;
    r.x = fmaxf(h.x, 0.f); r.y = fmaxf(h.y, 0.f);
    r.z = fmaxf(h.z, 0.f); r.w = fmaxf(h.w, 0.f);
    if (wz) {
        unsigned int lo = (unsigned int)f2bf(r.x) | ((unsigned int)f2bf(r.y) << 16);
        unsigned int hi = (unsigned int)f2bf(r.z) | ((unsigned int)f2bf(r.w) << 16);
        uint2 p; p.x = lo; p.y = hi;
        ((uint2*)ZP)[e] = p;
    }
    float4 x = ((const float4*)XD)[e];
    r.x += x.x; r.y += x.y; r.z += x.z; r.w += x.w;
    ((float4*)Hb)[e] = r;
}

// z_new = relu(H); res = z_new - z_prev; accumulate per-encode sums; H <- z_new
__global__ void kFinalRes(float* __restrict__ Hb, const unsigned short* __restrict__ ZP,
                          float* __restrict__ acc) {
    __shared__ float sR[8], sZ[8];
    size_t e = (size_t)blockIdx.x * 256 + threadIdx.x;
    int p = (int)(blockIdx.x >> 13);                 // 8192 blocks per encode
    float4 h = ((const float4*)Hb)[e];
    float4 z;
    z.x = fmaxf(h.x, 0.f); z.y = fmaxf(h.y, 0.f);
    z.z = fmaxf(h.z, 0.f); z.w = fmaxf(h.w, 0.f);
    uint2 q = ((const uint2*)ZP)[e];
    float4 zp;
    zp.x = bf2f(q.x & 0xffffu); zp.y = bf2f(q.x >> 16);
    zp.z = bf2f(q.y & 0xffffu); zp.w = bf2f(q.y >> 16);
    float rx = z.x - zp.x, ry = z.y - zp.y, rz = z.z - zp.z, rw = z.w - zp.w;
    float sr = rx * rx + ry * ry + rz * rz + rw * rw;
    float sz = z.x * z.x + z.y * z.y + z.z * z.z + z.w * z.w;
    ((float4*)Hb)[e] = z;
    for (int off = 16; off; off >>= 1) { sr += __shfl_xor(sr, off); sz += __shfl_xor(sz, off); }
    int w = threadIdx.x >> 5, lane = threadIdx.x & 31;
    if (!lane) { sR[w] = sr; sZ[w] = sz; }
    __syncthreads();
    if (threadIdx.x == 0) {
        float tr = 0.f, tz = 0.f;
        for (int i = 0; i < 8; i++) { tr += sR[i]; tz += sZ[i]; }
        atomicAdd(&acc[p], tr);
        atomicAdd(&acc[2 + p], tz);
    }
}

// masked mean-pool over l: one wave per (b,i)
__global__ void kPool(const float* __restrict__ Hb, const int* __restrict__ x1,
                      const int* __restrict__ x2, float* __restrict__ meanY) {
    int w = threadIdx.x >> 5, lane = threadIdx.x & 31;
    int row = blockIdx.x * 8 + w;                    // 8192 rows
    int b = row >> 8, i = row & 255;
    int p = b >> 4, bb = b & 15;
    const int* T = p ? x2 : x1;
    const float4* hv = (const float4*)(Hb + ((size_t)b * H_MODEL + i) * SEQ);
    const int4* tv = (const int4*)(T + bb * SEQ);
    float sy = 0.f, sm = 0.f;
    for (int k = 0; k < 16; k++) {
        float4 f = hv[lane + k * 32];
        int4 tq = tv[lane + k * 32];
        sy += f.x * (tq.x != 0) + f.y * (tq.y != 0) + f.z * (tq.z != 0) + f.w * (tq.w != 0);
        sm += (float)((tq.x != 0) + (tq.y != 0) + (tq.z != 0) + (tq.w != 0));
    }
    for (int off = 16; off; off >>= 1) { sy += __shfl_xor(sy, off); sm += __shfl_xor(sm, off); }
    if (!lane) meanY[((size_t)p * 16 + bb) * 256 + i] = sy / (sm + 1e-8f);
}

// ------------------------------- MLP head ----------------------------------

__global__ void kFc1(const float* __restrict__ meanY, const float* __restrict__ w,
                     const float* __restrict__ bias, float* __restrict__ hfc) {
    int b = blockIdx.x, i = threadIdx.x;
    const float* y1 = meanY + (size_t)b * 256;
    const float* y2 = meanY + (size_t)(16 + b) * 256;
    float a = bias[i];
    for (int k = 0; k < 256; k++) {
        float u = y1[k], v = y2[k];
        a = fmaf(u,            w[(size_t)k * 256 + i], a);
        a = fmaf(v,            w[(size_t)(256 + k) * 256 + i], a);
        a = fmaf(fabsf(u - v), w[(size_t)(512 + k) * 256 + i], a);
        a = fmaf(u * v,        w[(size_t)(768 + k) * 256 + i], a);
    }
    float x = a;
    hfc[b * 256 + i] = 0.5f * x * (1.f + tanhf(0.7978845608f * (x + 0.044715f * x * x * x)));
}

__global__ void kFc2Fin(const float* __restrict__ hfc, const float* __restrict__ w,
                        const float* __restrict__ bias, const float* __restrict__ acc,
                        float* __restrict__ out) {
    int t = threadIdx.x;
    if (t < 32) {
        int b = t >> 1, c = t & 1;
        float s = bias[c];
        for (int i = 0; i < 256; i++) s = fmaf(hfc[b * 256 + i], w[i * 2 + c], s);
        out[t] = s;
    } else if (t == 32) {
        out[32] = (acc[0] + acc[1]) * (0.5f / 8388608.f);   // mean(res^2) avg of encodes
    } else if (t == 33) {
        out[33] = 4.f;                                      // n_steps
    } else if (t == 34) {
        float z0 = sqrtf(acc[0]) / (sqrtf(acc[2]) + 1e-8f);
        float z1 = sqrtf(acc[1]) / (sqrtf(acc[3]) + 1e-8f);
        out[34] = 0.5f * (z0 + z1);
    }
}

// ------------------------------- launcher ----------------------------------

extern "C" void kernel_launch(void* const* d_in, const int* in_sizes, int n_in,
                              void* d_out, int out_size, void* d_ws, size_t ws_size,
                              hipStream_t stream) {
    (void)in_sizes; (void)n_in; (void)out_size; (void)ws_size;
    const int*   x1   = (const int*)d_in[0];
    const int*   x2   = (const int*)d_in[1];
    const float* emb  = (const float*)d_in[2];
    const float* elnw = (const float*)d_in[3];
    const float* elnb = (const float*)d_in[4];
    const float* fc1w = (const float*)d_in[65];
    const float* fc1b = (const float*)d_in[66];
    const float* fc2w = (const float*)d_in[67];
    const float* fc2b = (const float*)d_in[68];
    auto BP = [&](int blk, int k) -> const float* { return (const float*)d_in[5 + blk * 10 + k]; };
    // per-block tensors: 0 log_dt, 1 A_re, 2 A_im, 3 C_re, 4 C_im, 5 D, 6 Wo, 7 bo, 8 ln_w, 9 ln_b

    char* wsp = (char*)d_ws;
    size_t o = 0;
    float*          ACC = (float*)(wsp + o);          o += 256;
    float4*         DP  = (float4*)(wsp + o);         o += (size_t)6 * 16384 * sizeof(float4);
    unsigned short* WT  = (unsigned short*)(wsp + o); o += (size_t)6 * 65536 * 2;
    float*          MY  = (float*)(wsp + o);          o += (size_t)2 * 16 * 256 * 4;
    float*          HFC = (float*)(wsp + o);          o += (size_t)16 * 256 * 4;
    o = (o + 255) & ~(size_t)255;
    float*          Hb  = (float*)(wsp + o);          o += NELEM * 4;
    float*          Z   = (float*)(wsp + o);          o += NELEM * 4;
    float*          XD  = (float*)(wsp + o);          o += NELEM * 4;
    unsigned short* YB  = (unsigned short*)(wsp + o); o += NELEM * 2;
    unsigned short* ZP  = (unsigned short*)(wsp + o); o += NELEM * 2;

    kInit<<<1, 64, 0, stream>>>(ACC);
    for (int blk = 0; blk < 6; blk++) {
        kPrepParams<<<64, 256, 0, stream>>>(BP(blk, 0), BP(blk, 1), BP(blk, 2),
                                            BP(blk, 3), BP(blk, 4), DP + (size_t)blk * 16384);
        kPrepW<<<256, 256, 0, stream>>>(BP(blk, 6), WT + (size_t)blk * 65536);
    }

    kEmbedLN<<<dim3(64, 32), 256, 0, stream>>>(x1, x2, emb, elnw, elnb, Hb);

    auto runBlock = [&](int blk) {
        kLN<<<dim3(8, 32), 256, 0, stream>>>(Hb, BP(blk, 8), BP(blk, 9), Z);
        kScan<<<128, 256, 0, stream>>>(DP + (size_t)blk * 16384, Z, YB, BP(blk, 5));
        kGemm<<<dim3(16, 2, 32), 256, 0, stream>>>(WT + (size_t)blk * 65536, YB, Hb, BP(blk, 7));
    };

    for (int blk = 0; blk < 4; blk++) runBlock(blk);          // 4 stacked blocks

    (void)hipMemcpyAsync(XD, Hb, NELEM * 4, hipMemcpyDeviceToDevice, stream);
    for (int it = 0; it < 4; it++) {                          // DEQ fixed-point
        if (it > 0) kReluAdd<<<16384, 256, 0, stream>>>(Hb, XD, ZP, (it == 3) ? 1 : 0);
        runBlock(4);
        runBlock(5);
    }
    kFinalRes<<<16384, 256, 0, stream>>>(Hb, ZP, ACC);
    kPool<<<1024, 256, 0, stream>>>(Hb, x1, x2, MY);
    kFc1<<<16, 256, 0, stream>>>(MY, fc1w, fc1b, HFC);
    kFc2Fin<<<1, 64, 0, stream>>>(HFC, fc2w, fc2b, ACC, (float*)d_out);
}
